// LSTM_37580963840285
// MI455X (gfx1250) — matrix operand order repaired and checked by reading, not compiled, not run on hardware
//
#include <hip/hip_runtime.h>
#include <hip/hip_bf16.h>

// ---------------------------------------------------------------------------
// 2-layer LSTM (T=512,B=64,IN=64,H=1024) for MI455X (gfx1250, wave32, WMMA).
// Persistent cooperative kernel: 64 workgroups x 512 threads (16 waves).
// Each workgroup owns 16 hidden units (all 4 gates); each wave computes one
// 16x16 gate tile via v_wmma_f32_16x16x32_bf16 over the K dimension, with a
// software-pipelined (2-deep, dual-accumulator) K loop and WGP-scope
// prefetch of the weight/activation streams.
// Weights pre-swizzled to per-lane WMMA B-fragment order (bf16, L2-resident).
// fp32 cell state kept in LDS across all 512 steps; grid barrier between
// layer-steps (2 per timestep).
// ---------------------------------------------------------------------------

#define T_SEQ 512
#define BATCH 64
#define IN_DIM 64
#define HID 1024
#define NWG 64            // workgroups; each owns 16 hidden units
#define NKT0 34           // K tiles layer0: (64+1024)/32
#define NKT1 64           // K tiles layer1: (1024+1024)/32

typedef __attribute__((ext_vector_type(8)))  float  v8f;
typedef __attribute__((ext_vector_type(16))) __bf16 v16bf;

// ---------------- helpers ----------------

__device__ __host__ inline __bf16 f2bf(float f) {
  unsigned u;
  __builtin_memcpy(&u, &f, 4);
  unsigned r = u + 0x7FFFu + ((u >> 16) & 1u);   // RNE
  unsigned short h = (unsigned short)(r >> 16);
  __bf16 out;
  __builtin_memcpy(&out, &h, 2);
  return out;
}

__device__ inline float fast_sigmoid(float x) { return 1.0f / (1.0f + __expf(-x)); }
__device__ inline float fast_tanh(float x)    { float e = __expf(2.0f * x); return 1.0f - 2.0f / (e + 1.0f); }

__device__ inline v16bf mk_v16bf(uint4 lo, uint4 hi) {
  union { uint4 q[2]; v16bf v; } u;
  u.q[0] = lo; u.q[1] = hi;
  return u.v;
}

// A fragment (16x32 bf16, per ISA layout): two 16B loads at p and p+32B,
// where p already includes row*stride + (lane>=16 ? 8 : 0).
__device__ inline v16bf ld_a(const __bf16* p) {
  uint4 lo = *reinterpret_cast<const uint4*>(p);
  uint4 hi = *reinterpret_cast<const uint4*>(p + 16);
  return mk_v16bf(lo, hi);
}

// B fragment: pre-swizzled, 32 contiguous bytes per lane (p includes lane*16).
__device__ inline v16bf ld_b(const __bf16* p) {
  uint4 lo = *reinterpret_cast<const uint4*>(p);
  uint4 hi = *reinterpret_cast<const uint4*>(p + 8);
  return mk_v16bf(lo, hi);
}

__device__ inline void prefetch_wgp(const void* p) {
  // default scope 0 = WGP: pulls the line through all cache levels
  asm volatile("global_prefetch_b8 %0, off"
               :: "v"((unsigned long long)(uintptr_t)p) : "memory");
}

// Software-pipelined GEMM over NKT contiguous K tiles (NKT even, >= 2).
// A tiles at abase + row*astride + kt*32 (+lane k-offset), B tiles at
// bbase + kt*512 (+lane*16).  Dual accumulator chains (even/odd tiles).
template <int NKT>
__device__ inline void gemm_region(v8f& accA, v8f& accB,
                                   const __bf16* __restrict__ abase, int astride,
                                   int row,
                                   const __bf16* __restrict__ bbase, int lane) {
  const __bf16* ap = abase + (size_t)row * astride + ((lane >> 4) << 3);
  const __bf16* bp = bbase + lane * 16;
  v16bf a0 = ld_a(ap +  0), b0 = ld_b(bp +   0);
  v16bf a1 = ld_a(ap + 32), b1 = ld_b(bp + 512);
#pragma unroll
  for (int kt = 0; kt < NKT; kt += 2) {
    prefetch_wgp(bp + (size_t)(kt + 6) * 512);
    prefetch_wgp(ap + (size_t)(kt + 6) * 32);
    // preload tiles kt+2, kt+3 (slack-padded buffers make overrun reads safe)
    v16bf a2 = ld_a(ap + (size_t)(kt + 2) * 32);
    v16bf b2 = ld_b(bp + (size_t)(kt + 2) * 512);
    v16bf a3 = ld_a(ap + (size_t)(kt + 3) * 32);
    v16bf b3 = ld_b(bp + (size_t)(kt + 3) * 512);
    accA = __builtin_amdgcn_wmma_f32_16x16x32_bf16(
        false, a0, false, b0, (short)0, accA, false, false);
    accB = __builtin_amdgcn_wmma_f32_16x16x32_bf16(
        false, a1, false, b1, (short)0, accB, false, false);
    a0 = a2; b0 = b2; a1 = a3; b1 = b3;
  }
}

// grid-wide sense-reversing barrier (all 64 WGs co-resident)
__device__ inline void grid_barrier(unsigned* cnt, unsigned* gen) {
  __syncthreads();
  if (threadIdx.x == 0) {
    __threadfence();
    unsigned g = __hip_atomic_load(gen, __ATOMIC_RELAXED, __HIP_MEMORY_SCOPE_AGENT);
    unsigned prev = __hip_atomic_fetch_add(cnt, 1u, __ATOMIC_ACQ_REL, __HIP_MEMORY_SCOPE_AGENT);
    if (prev == gridDim.x - 1) {
      __hip_atomic_store(cnt, 0u, __ATOMIC_RELAXED, __HIP_MEMORY_SCOPE_AGENT);
      __hip_atomic_store(gen, g + 1u, __ATOMIC_RELEASE, __HIP_MEMORY_SCOPE_AGENT);
    } else {
      while (__hip_atomic_load(gen, __ATOMIC_ACQUIRE, __HIP_MEMORY_SCOPE_AGENT) == g)
        __builtin_amdgcn_s_sleep(1);
    }
    __threadfence();
  }
  __syncthreads();
}

// activation + LDS gate exchange + cell update for this WG's 16 hidden units
__device__ inline void gates_update(v8f acc, int gt, int mt, int lane,
                                    float* gatesLDS, float* cst,
                                    __bf16* __restrict__ hw,
                                    float* __restrict__ outrow, int wg) {
  int n = lane & 15;
  int mhi = (lane >> 4) << 3;
  __syncthreads();
#pragma unroll
  for (int r = 0; r < 8; ++r) {
    int m = mt * 16 + mhi + r;
    float v = acc[r];
    v = (gt == 2) ? fast_tanh(v) : fast_sigmoid(v);
    gatesLDS[(gt * 64 + m) * 16 + n] = v;
  }
  __syncthreads();
  if (gt == 0) {
#pragma unroll
    for (int r = 0; r < 8; ++r) {
      int m = mt * 16 + mhi + r;
      float iv = gatesLDS[(0 * 64 + m) * 16 + n];
      float fv = gatesLDS[(1 * 64 + m) * 16 + n];
      float gv = gatesLDS[(2 * 64 + m) * 16 + n];
      float ov = gatesLDS[(3 * 64 + m) * 16 + n];
      float c = fv * cst[m * 16 + n] + iv * gv;
      cst[m * 16 + n] = c;
      float h = ov * fast_tanh(c);
      hw[(size_t)m * HID + wg * 16 + n] = f2bf(h);
      if (outrow) outrow[(size_t)m * HID + wg * 16 + n] = h;
    }
  }
}

// ---------------- kernels ----------------

// biases, x->bf16, zero hidden double-buffers, zero barrier state
__global__ void prep_kernel(const float* __restrict__ x,
                            const float* __restrict__ bih0, const float* __restrict__ bhh0,
                            const float* __restrict__ bih1, const float* __restrict__ bhh1,
                            __bf16* __restrict__ xb,
                            float* __restrict__ bias0, float* __restrict__ bias1,
                            __bf16* __restrict__ h0, __bf16* __restrict__ h1,
                            unsigned* __restrict__ bar) {
  size_t id = (size_t)blockIdx.x * blockDim.x + threadIdx.x;
  if (id < (size_t)T_SEQ * BATCH * IN_DIM) xb[id] = f2bf(x[id]);
  if (id < 4 * HID) {
    bias0[id] = bih0[id] + bhh0[id];
    bias1[id] = bih1[id] + bhh1[id];
  }
  if (id < 2 * (size_t)BATCH * HID) {
    unsigned short z = 0; __bf16 zb; __builtin_memcpy(&zb, &z, 2);
    h0[id] = zb; h1[id] = zb;
  }
  if (id < 2) bar[id] = 0u;
}

// Swizzle [W_ih | W_hh] (row-major fp32) into per-lane WMMA B-fragment order:
// dst[(((wg*4+gt)*NKT + kt)*32 + lane)*16 + e] =
//   Wcat[gt*H + wg*16 + lane%16][kt*32 + (e%8) + (lane/16)*8 + (e/8)*16]
__global__ void pack_kernel(const float* __restrict__ W_ih, const float* __restrict__ W_hh,
                            __bf16* __restrict__ dst, int KI, int KH, int NKT) {
  size_t id = (size_t)blockIdx.x * blockDim.x + threadIdx.x;
  size_t total = (size_t)NWG * 4 * NKT * 512;
  if (id >= total) return;
  int e = (int)(id & 15);
  int lane = (int)((id >> 4) & 31);
  size_t tile = id >> 9;
  int kt = (int)(tile % NKT);
  size_t rest = tile / NKT;
  int gt = (int)(rest & 3);
  int wg = (int)(rest >> 2);
  int g_col = gt * HID + wg * 16 + (lane & 15);
  int k = kt * 32 + (e & 7) + ((lane >> 4) << 3) + ((e >> 3) << 4);
  float v = (k < KI) ? W_ih[(size_t)g_col * KI + k]
                     : W_hh[(size_t)g_col * KH + (k - KI)];
  dst[id] = f2bf(v);
}

__global__ void __launch_bounds__(512)
lstm_persistent(const __bf16* __restrict__ xb,
                const __bf16* __restrict__ pack0, const __bf16* __restrict__ pack1,
                const float* __restrict__ bias0, const float* __restrict__ bias1,
                __bf16* __restrict__ h0buf, __bf16* __restrict__ h1buf,
                float* __restrict__ out,
                unsigned* __restrict__ bar_cnt, unsigned* __restrict__ bar_gen) {
  __shared__ float gatesLDS[4 * 64 * 16];   // 16 KB
  __shared__ float cst0[64 * 16];           // 4 KB fp32 cell state, layer0
  __shared__ float cst1[64 * 16];           // 4 KB fp32 cell state, layer1

  const int tid  = threadIdx.x;
  const int wg   = blockIdx.x;
  const int wave = tid >> 5;
  const int lane = tid & 31;
  const int mt   = wave >> 2;     // batch tile 0..3
  const int gt   = wave & 3;      // gate i,f,g,o
  const int mrow = mt * 16 + (lane & 15);

  for (int i = tid; i < 64 * 16; i += blockDim.x) { cst0[i] = 0.f; cst1[i] = 0.f; }
  __syncthreads();

  const float b0 = bias0[gt * HID + wg * 16 + (lane & 15)];
  const float b1 = bias1[gt * HID + wg * 16 + (lane & 15)];
  const size_t HB = (size_t)BATCH * HID;
  const int tb0 = (wg * 4 + gt) * NKT0;
  const int tb1 = (wg * 4 + gt) * NKT1;

  for (int t = 0; t < T_SEQ; ++t) {
    const __bf16* h0r = h0buf + (size_t)(t & 1) * HB;
    __bf16*       h0w = h0buf + (size_t)((t + 1) & 1) * HB;
    const __bf16* h1r = h1buf + (size_t)(t & 1) * HB;
    __bf16*       h1w = h1buf + (size_t)((t + 1) & 1) * HB;

    // ---- layer 0: gates = [x_t | h0] @ Wcat0^T + bias ----
    {
      v8f accA = {b0, b0, b0, b0, b0, b0, b0, b0};
      v8f accB = {0.f, 0.f, 0.f, 0.f, 0.f, 0.f, 0.f, 0.f};
      gemm_region<2>(accA, accB, xb + (size_t)t * BATCH * IN_DIM, IN_DIM, mrow,
                     pack0 + (size_t)tb0 * 512, lane);
      gemm_region<32>(accA, accB, h0r, HID, mrow,
                      pack0 + (size_t)(tb0 + 2) * 512, lane);
      gates_update(accA + accB, gt, mt, lane, gatesLDS, cst0, h0w, nullptr, wg);
    }
    grid_barrier(bar_cnt, bar_gen);

    // ---- layer 1: gates = [h0_new | h1] @ Wcat1^T + bias ----
    {
      v8f accA = {b1, b1, b1, b1, b1, b1, b1, b1};
      v8f accB = {0.f, 0.f, 0.f, 0.f, 0.f, 0.f, 0.f, 0.f};
      gemm_region<32>(accA, accB, h0w, HID, mrow,
                      pack1 + (size_t)tb1 * 512, lane);
      gemm_region<32>(accA, accB, h1r, HID, mrow,
                      pack1 + (size_t)(tb1 + 32) * 512, lane);
      gates_update(accA + accB, gt, mt, lane, gatesLDS, cst1, h1w,
                   out + (size_t)t * HB, wg);
    }
    grid_barrier(bar_cnt, bar_gen);
  }
}

// y[b] = lstm_out[T-1, b, :] . W_lin[0, :] + b_lin[0]
__global__ void final_linear(const float* __restrict__ out,
                             const float* __restrict__ W_lin,
                             const float* __restrict__ b_lin,
                             float* __restrict__ y) {
  __shared__ float red[256];
  int b = blockIdx.x;
  const float* row = out + ((size_t)(T_SEQ - 1) * BATCH + b) * HID;
  float s = 0.f;
  for (int h = threadIdx.x; h < HID; h += 256) s += row[h] * W_lin[h];
  red[threadIdx.x] = s;
  __syncthreads();
  for (int off = 128; off; off >>= 1) {
    if (threadIdx.x < off) red[threadIdx.x] += red[threadIdx.x + off];
    __syncthreads();
  }
  if (threadIdx.x == 0) y[b] = red[0] + b_lin[0];
}

// ---------------- host ----------------

extern "C" void kernel_launch(void* const* d_in, const int* in_sizes, int n_in,
                              void* d_out, int out_size, void* d_ws, size_t ws_size,
                              hipStream_t stream) {
  (void)in_sizes; (void)n_in; (void)out_size; (void)ws_size;
  const float* x     = (const float*)d_in[0];
  const float* W_ih0 = (const float*)d_in[1];
  const float* W_hh0 = (const float*)d_in[2];
  const float* b_ih0 = (const float*)d_in[3];
  const float* b_hh0 = (const float*)d_in[4];
  const float* W_ih1 = (const float*)d_in[5];
  const float* W_hh1 = (const float*)d_in[6];
  const float* b_ih1 = (const float*)d_in[7];
  const float* b_hh1 = (const float*)d_in[8];
  const float* W_lin = (const float*)d_in[9];
  const float* b_lin = (const float*)d_in[10];
  float* out = (float*)d_out;

  const size_t SLACK = 16384;   // lookahead slack for pipelined loads/prefetch
  char* base = (char*)d_ws;
  size_t off = 0;
  auto take = [&](size_t bytes) -> char* {
    char* p = base + off;
    off = (off + bytes + 255) & ~(size_t)255;
    return p;
  };
  __bf16*   pack0 = (__bf16*)take((size_t)NWG * 4 * NKT0 * 512 * 2 + SLACK);
  __bf16*   pack1 = (__bf16*)take((size_t)NWG * 4 * NKT1 * 512 * 2 + SLACK);
  __bf16*   xb    = (__bf16*)take((size_t)T_SEQ * BATCH * IN_DIM * 2 + SLACK);
  float*    bias0 = (float*)take(4 * HID * 4);
  float*    bias1 = (float*)take(4 * HID * 4);
  __bf16*   h0buf = (__bf16*)take(2 * (size_t)BATCH * HID * 2 + SLACK);
  __bf16*   h1buf = (__bf16*)take(2 * (size_t)BATCH * HID * 2 + SLACK);
  unsigned* bar   = (unsigned*)take(256);

  prep_kernel<<<(T_SEQ * BATCH * IN_DIM) / 256, 256, 0, stream>>>(
      x, b_ih0, b_hh0, b_ih1, b_hh1, xb, bias0, bias1, h0buf, h1buf, bar);

  pack_kernel<<<((size_t)NWG * 4 * NKT0 * 512) / 256, 256, 0, stream>>>(
      W_ih0, W_hh0, pack0, IN_DIM, HID, NKT0);
  pack_kernel<<<((size_t)NWG * 4 * NKT1 * 512) / 256, 256, 0, stream>>>(
      W_ih1, W_hh1, pack1, HID, HID, NKT1);

  lstm_persistent<<<NWG, 512, 0, stream>>>(
      xb, pack0, pack1, bias0, bias1, h0buf, h1buf, out, bar, bar + 1);

  final_linear<<<BATCH, 256, 0, stream>>>(
      out, W_lin, b_lin, out + (size_t)T_SEQ * BATCH * HID);
}
